// GraphWriter_69655779606779
// MI455X (gfx1250) — compile-verified, matrix operand order checked by reading, not run on hardware
//
#include <hip/hip_runtime.h>
#include <hip/hip_bf16.h>
#include <math.h>

// ---------------- problem constants ----------------
#define NN_NODES 50000
#define FEAT     256
#define HEADS    4
#define HDIM     64
#define EDGES    800000
#define QKV_N    768          // q|k|v concatenated columns
#define LN_EPS   1e-5f

typedef unsigned short b16u;
typedef __bf16 v16bf __attribute__((ext_vector_type(16)));
typedef float  v8f   __attribute__((ext_vector_type(8)));

struct Frag {
  union {
    v16bf bf;
    uint4 q[2];
  };
};

__device__ __forceinline__ b16u f32_to_bf16(float f) {
  unsigned u = __float_as_uint(f);
  unsigned r = u + 0x7FFFu + ((u >> 16) & 1u);   // round-to-nearest-even
  return (b16u)(r >> 16);
}

__device__ __forceinline__ v8f wmma_bf16(const Frag& a, const Frag& b, v8f c) {
  return __builtin_amdgcn_wmma_f32_16x16x32_bf16(
      /*neg_a=*/false, a.bf, /*neg_b=*/false, b.bf,
      /*c_mod=*/(short)0, c, /*reuse_a=*/false, /*reuse_b=*/false);
}

// CDNA5 async copy: global -> LDS, 16B per active lane, tracked by ASYNCcnt.
__device__ __forceinline__ void async_ld_b128(unsigned lds_addr, const void* gaddr) {
  asm volatile("global_load_async_to_lds_b128 %0, %1, off"
               :: "v"(lds_addr), "v"(gaddr) : "memory");
}
__device__ __forceinline__ void wait_async0() {
  asm volatile("s_wait_asynccnt 0x0" ::: "memory");
}

// ---------------- packing kernels ----------------
__global__ void pack_feat_kernel(const float* __restrict__ feat, b16u* __restrict__ out) {
  int tid = blockIdx.x * blockDim.x + threadIdx.x;
  if (tid < NN_NODES * FEAT) out[tid] = f32_to_bf16(feat[tid]);
}

// Build transposed bf16 weights: Bt[n][k] = W[k][n]
__global__ void pack_weights_kernel(const float* __restrict__ Wq, const float* __restrict__ Wk,
                                    const float* __restrict__ Wv, const float* __restrict__ W1,
                                    const float* __restrict__ W2,
                                    b16u* __restrict__ Wqkvt,  // [768][256]
                                    b16u* __restrict__ W1t,    // [1024][256]
                                    b16u* __restrict__ W2t) {  // [256][1024]
  int tid = blockIdx.x * blockDim.x + threadIdx.x;
  if (tid < 768 * 256) {
    int n = tid >> 8, k = tid & 255;
    const float* W = (n < 256) ? Wq : ((n < 512) ? Wk : Wv);
    int nn = n & 255;
    Wqkvt[tid] = f32_to_bf16(W[k * 256 + nn]);
  }
  if (tid < 1024 * 256) {              // W1 is [256][1024]
    int n = tid >> 8, k = tid & 255;
    W1t[tid] = f32_to_bf16(W1[k * 1024 + n]);
  }
  if (tid < 256 * 1024) {              // W2 is [1024][256]
    int n = tid >> 10, k = tid & 1023;
    W2t[tid] = f32_to_bf16(W2[k * 256 + n]);
  }
}

__global__ void init_ws_kernel(float* __restrict__ emax, float* __restrict__ denom,
                               float* __restrict__ ft2) {
  int tid = blockIdx.x * blockDim.x + threadIdx.x;
  if (tid < NN_NODES * HEADS) {
    emax[tid]  = -INFINITY;
    denom[tid] = 0.0f;
  }
  if (tid < NN_NODES * FEAT) ft2[tid] = 0.0f;
}

// ---------------- QKV GEMM: [50000,256] x [256,768] -> [50000,768] ----------------
// One block per 16-row stripe. A tile staged in LDS via async copy; each wave
// keeps its 8 A-fragments in registers and register-blocks 6 N-tiles (48 wmma/wave).
#define APAD 8
__global__ __launch_bounds__(256) void qkv_gemm_kernel(const b16u* __restrict__ A,
                                                       const b16u* __restrict__ Bt,
                                                       float* __restrict__ C) {
  __shared__ b16u Atile[16][256 + APAD];
  const int K = 256;
  int tid = threadIdx.x;
  int lane = tid & 31, w = tid >> 5;
  int m0 = blockIdx.x * 16;
  int r = lane & 15, sel = lane >> 4;
  int col = lane & 15, rowoff = sel * 8;

  // stage A tile (16 x 256 bf16 = 512 x 16B chunks, 2 per thread)
#pragma unroll
  for (int j = 0; j < 2; ++j) {
    int ci = tid + j * 256;               // 0..511
    int row = ci >> 5, cc = (ci & 31) * 8;
    async_ld_b128((unsigned)(uintptr_t)&Atile[row][cc],
                  A + (size_t)(m0 + row) * K + cc);
  }
  wait_async0();
  __syncthreads();

  // load all 8 A-fragments once
  Frag fa[8];
#pragma unroll
  for (int s = 0; s < 8; ++s) {
    int ka = s * 32 + sel * 8;
    fa[s].q[0] = *(const uint4*)&Atile[r][ka];
    fa[s].q[1] = *(const uint4*)&Atile[r][ka + 16];
  }

  // 6 N-tiles per wave (8 waves x 6 x 16 = 768 columns)
  for (int t = 0; t < 6; ++t) {
    int n0 = (w * 6 + t) * 16;
    const b16u* brow = Bt + (size_t)(n0 + r) * K;
    v8f acc = {};
#pragma unroll
    for (int s = 0; s < 8; ++s) {
      int ka = s * 32 + sel * 8;
      Frag fb;
      fb.q[0] = *(const uint4*)(brow + ka);
      fb.q[1] = *(const uint4*)(brow + ka + 16);
      acc = wmma_bf16(fa[s], fb, acc);
    }
    float* crow = C + (size_t)(m0 + rowoff) * QKV_N + n0 + col;
#pragma unroll
    for (int i = 0; i < 8; ++i) crow[(size_t)i * QKV_N] = acc[i];
  }
}

// ---------------- edge kernels ----------------
// qkv layout per node: [0..255]=q, [256..511]=k, [512..767]=v
__global__ void edge_score_kernel(const float* __restrict__ qkv, const int* __restrict__ src,
                                  const int* __restrict__ dst, float* __restrict__ ebuf,
                                  float* __restrict__ emax) {
  int tid = blockIdx.x * blockDim.x + threadIdx.x;
  if (tid >= EDGES * HEADS) return;
  int e = tid >> 2, h = tid & 3;
  int s = src[e], d = dst[e];
  const float4* kp = (const float4*)(qkv + (size_t)s * QKV_N + 256 + h * HDIM);
  const float4* qp = (const float4*)(qkv + (size_t)d * QKV_N + h * HDIM);
  float acc = 0.0f;
#pragma unroll
  for (int i = 0; i < 16; ++i) {
    float4 a = kp[i], b = qp[i];
    acc += a.x * b.x + a.y * b.y + a.z * b.z + a.w * b.w;
  }
  acc *= (1.0f / 16.0f);  // 1/sqrt(H*D) = 1/16
  ebuf[tid] = acc;
  int* addr = (int*)(emax + (size_t)d * HEADS + h);
  if (acc >= 0.0f) atomicMax(addr, __float_as_int(acc));
  else             atomicMin((unsigned int*)addr, __float_as_uint(acc));
}

__global__ void edge_exp_kernel(const int* __restrict__ dst, const float* __restrict__ emax,
                                float* __restrict__ ebuf, float* __restrict__ denom) {
  int tid = blockIdx.x * blockDim.x + threadIdx.x;
  if (tid >= EDGES * HEADS) return;
  int e = tid >> 2, h = tid & 3;
  int d = dst[e];
  float m  = emax[(size_t)d * HEADS + h];
  float ex = __expf(ebuf[tid] - m);
  ebuf[tid] = ex;
  atomicAdd(&denom[(size_t)d * HEADS + h], ex);
}

__global__ void edge_agg_kernel(const float* __restrict__ qkv, const int* __restrict__ src,
                                const int* __restrict__ dst, const float* __restrict__ ebuf,
                                const float* __restrict__ denom, float* __restrict__ ft2) {
  int tid = blockIdx.x * blockDim.x + threadIdx.x;
  if (tid >= EDGES * HEADS) return;
  int e = tid >> 2, h = tid & 3;
  int s = src[e], d = dst[e];
  float coef = ebuf[tid] / denom[(size_t)d * HEADS + h];
  const float* vp = qkv + (size_t)s * QKV_N + 512 + h * HDIM;
  float* op = ft2 + (size_t)d * FEAT + h * HDIM;
#pragma unroll 4
  for (int i = 0; i < HDIM; ++i) atomicAdd(&op[i], coef * vp[i]);
}

// ---------------- residual + LayerNorm (one wave32 per row) ----------------
__global__ void residual_ln_kernel(const float* __restrict__ ft2, const float* __restrict__ feat,
                                   const float* __restrict__ g, const float* __restrict__ b,
                                   float* __restrict__ yf, b16u* __restrict__ ybf) {
  int wave = (blockIdx.x * blockDim.x + threadIdx.x) >> 5;
  int lane = threadIdx.x & 31;
  if (wave >= NN_NODES) return;
  const float* a = ft2  + (size_t)wave * FEAT;
  const float* f = feat + (size_t)wave * FEAT;
  float x[8];
  float s = 0.0f;
#pragma unroll
  for (int i = 0; i < 8; ++i) {
    int c = lane + 32 * i;
    x[i] = a[c] + f[c];
    s += x[i];
  }
#pragma unroll
  for (int off = 16; off > 0; off >>= 1) s += __shfl_xor(s, off, 32);
  float mu = s * (1.0f / FEAT);
  float v = 0.0f;
#pragma unroll
  for (int i = 0; i < 8; ++i) { float dv = x[i] - mu; v += dv * dv; }
#pragma unroll
  for (int off = 16; off > 0; off >>= 1) v += __shfl_xor(v, off, 32);
  float rs = rsqrtf(v * (1.0f / FEAT) + LN_EPS);
#pragma unroll
  for (int i = 0; i < 8; ++i) {
    int c = lane + 32 * i;
    float yv = (x[i] - mu) * rs * g[c] + b[c];
    yf[(size_t)wave * FEAT + c]  = yv;
    ybf[(size_t)wave * FEAT + c] = f32_to_bf16(yv);
  }
}

// ---------------- fused FFN: GEMM1 + PReLU -> LDS -> GEMM2 + residual + LN ----------------
#define HPAD 8
__global__ __launch_bounds__(256) void ffn_fused_kernel(
    const b16u* __restrict__ Ybf, const float* __restrict__ Yf,
    const b16u* __restrict__ W1t, const b16u* __restrict__ W2t,
    const float* __restrict__ b1, const float* __restrict__ alpha,
    const float* __restrict__ b2, const float* __restrict__ g,
    const float* __restrict__ b, float* __restrict__ out) {
  __shared__ b16u  Atile[16][256 + APAD];  // staged LN(y) bf16 tile
  __shared__ b16u  hid[16][1024 + HPAD];   // bf16 hidden tile
  __shared__ float st[16][FEAT + 4];       // f32 pre-LN tile
  __shared__ float smu[16], srs[16];
  int tid  = threadIdx.x;
  int lane = tid & 31, w = tid >> 5;
  int m0 = blockIdx.x * 16;
  int r = lane & 15, sel = lane >> 4;
  int col = lane & 15, rowoff = sel * 8;

  // stage A tile via async copy
#pragma unroll
  for (int j = 0; j < 2; ++j) {
    int ci = tid + j * 256;
    int row = ci >> 5, cc = (ci & 31) * 8;
    async_ld_b128((unsigned)(uintptr_t)&Atile[row][cc],
                  Ybf + (size_t)(m0 + row) * FEAT + cc);
  }
  wait_async0();
  __syncthreads();

  // A fragments for GEMM1 (reused across all 8 N-tiles)
  Frag fa[8];
#pragma unroll
  for (int s = 0; s < 8; ++s) {
    int ka = s * 32 + sel * 8;
    fa[s].q[0] = *(const uint4*)&Atile[r][ka];
    fa[s].q[1] = *(const uint4*)&Atile[r][ka + 16];
  }

  // GEMM1: [16,256] x [256,1024]; 8 n-tiles per wave
  for (int t = 0; t < 8; ++t) {
    int n0 = (w * 8 + t) * 16;
    const b16u* brow = W1t + (size_t)(n0 + r) * FEAT;
    v8f acc = {};
#pragma unroll
    for (int s = 0; s < 8; ++s) {
      int ka = s * 32 + sel * 8;
      Frag fb;
      fb.q[0] = *(const uint4*)(brow + ka);
      fb.q[1] = *(const uint4*)(brow + ka + 16);
      acc = wmma_bf16(fa[s], fb, acc);
    }
    int n = n0 + col;
    float bias = b1[n], al = alpha[n];
#pragma unroll
    for (int i = 0; i < 8; ++i) {
      float x = acc[i] + bias;
      x = x > 0.0f ? x : al * x;    // PReLU
      hid[rowoff + i][n] = f32_to_bf16(x);
    }
  }
  __syncthreads();

  // GEMM2: [16,1024] x [1024,256]; 2 n-tiles per wave, shared A-fragment per k-step
  {
    int n0a = (w * 2) * 16, n0b = n0a + 16;
    const b16u* browa = W2t + (size_t)(n0a + r) * 1024;
    const b16u* browb = W2t + (size_t)(n0b + r) * 1024;
    v8f acc0 = {}, acc1 = {};
#pragma unroll 4
    for (int k0 = 0; k0 < 1024; k0 += 32) {
      int ka = k0 + sel * 8;
      Frag f2, fb0, fb1;
      f2.q[0]  = *(const uint4*)&hid[r][ka];
      f2.q[1]  = *(const uint4*)&hid[r][ka + 16];
      fb0.q[0] = *(const uint4*)(browa + ka);
      fb0.q[1] = *(const uint4*)(browa + ka + 16);
      fb1.q[0] = *(const uint4*)(browb + ka);
      fb1.q[1] = *(const uint4*)(browb + ka + 16);
      acc0 = wmma_bf16(f2, fb0, acc0);
      acc1 = wmma_bf16(f2, fb1, acc1);
    }
    int na = n0a + col, nb = n0b + col;
    float ba = b2[na], bb = b2[nb];
#pragma unroll
    for (int i = 0; i < 8; ++i) {
      int row = rowoff + i;
      st[row][na] = acc0[i] + ba + Yf[(size_t)(m0 + row) * FEAT + na];
      st[row][nb] = acc1[i] + bb + Yf[(size_t)(m0 + row) * FEAT + nb];
    }
  }
  __syncthreads();

  // per-row LayerNorm statistics
  if (tid < 16) {
    float s = 0.0f;
    for (int j = 0; j < FEAT; ++j) s += st[tid][j];
    float mu = s * (1.0f / FEAT);
    float v = 0.0f;
    for (int j = 0; j < FEAT; ++j) { float d = st[tid][j] - mu; v += d * d; }
    smu[tid] = mu;
    srs[tid] = rsqrtf(v * (1.0f / FEAT) + LN_EPS);
  }
  __syncthreads();

  int row = tid >> 4;
  int c0  = (tid & 15) * 16;
  float mu = smu[row], rs = srs[row];
#pragma unroll
  for (int j = 0; j < 16; ++j) {
    int c = c0 + j;
    out[(size_t)(m0 + row) * FEAT + c] = (st[row][c] - mu) * rs * g[c] + b[c];
  }
}

// ---------------- host launcher ----------------
extern "C" void kernel_launch(void* const* d_in, const int* in_sizes, int n_in,
                              void* d_out, int out_size, void* d_ws, size_t ws_size,
                              hipStream_t stream) {
  const float* feat  = (const float*)d_in[0];
  const int*   src   = (const int*)d_in[1];
  const int*   dst   = (const int*)d_in[2];
  const float* Wq    = (const float*)d_in[3];
  const float* Wk    = (const float*)d_in[4];
  const float* Wv    = (const float*)d_in[5];
  const float* ln_g  = (const float*)d_in[6];
  const float* ln_b  = (const float*)d_in[7];
  const float* W1    = (const float*)d_in[8];
  const float* b1    = (const float*)d_in[9];
  const float* alpha = (const float*)d_in[10];
  const float* W2    = (const float*)d_in[11];
  const float* b2    = (const float*)d_in[12];
  float* out = (float*)d_out;

  // workspace layout (all 256B aligned)
  char* ws = (char*)d_ws;
  size_t off = 0;
  auto take = [&](size_t bytes) {
    void* p = ws + off;
    off += (bytes + 255) & ~(size_t)255;
    return p;
  };
  b16u*  featb  = (b16u*)take((size_t)NN_NODES * FEAT * 2);
  b16u*  Wqkvt  = (b16u*)take((size_t)768 * 256 * 2);
  b16u*  W1t    = (b16u*)take((size_t)1024 * 256 * 2);
  b16u*  W2t    = (b16u*)take((size_t)256 * 1024 * 2);
  float* qkv    = (float*)take((size_t)NN_NODES * QKV_N * 4);
  float* ebuf   = (float*)take((size_t)EDGES * HEADS * 4);
  float* emax   = (float*)take((size_t)NN_NODES * HEADS * 4);
  float* denom  = (float*)take((size_t)NN_NODES * HEADS * 4);
  float* ft2    = (float*)take((size_t)NN_NODES * FEAT * 4);
  float* yf     = (float*)take((size_t)NN_NODES * FEAT * 4);
  b16u*  ybf    = (b16u*)take((size_t)NN_NODES * FEAT * 2);
  (void)ws_size; (void)n_in; (void)in_sizes; (void)out_size;

  const int T = 256;
  // 1. pack inputs to bf16
  pack_feat_kernel<<<(NN_NODES * FEAT + T - 1) / T, T, 0, stream>>>(feat, featb);
  pack_weights_kernel<<<(1024 * 256 + T - 1) / T, T, 0, stream>>>(Wq, Wk, Wv, W1, W2,
                                                                  Wqkvt, W1t, W2t);
  init_ws_kernel<<<(NN_NODES * FEAT + T - 1) / T, T, 0, stream>>>(emax, denom, ft2);

  // 2. QKV projection (WMMA + async LDS staging)
  qkv_gemm_kernel<<<NN_NODES / 16, T, 0, stream>>>(featb, Wqkvt, qkv);

  // 3. edge attention
  int egrid = (EDGES * HEADS + T - 1) / T;
  edge_score_kernel<<<egrid, T, 0, stream>>>(qkv, src, dst, ebuf, emax);
  edge_exp_kernel<<<egrid, T, 0, stream>>>(dst, emax, ebuf, denom);
  edge_agg_kernel<<<egrid, T, 0, stream>>>(qkv, src, dst, ebuf, denom, ft2);

  // 4. residual + LN
  residual_ln_kernel<<<(NN_NODES * 32 + T - 1) / T, T, 0, stream>>>(ft2, feat, ln_g, ln_b,
                                                                    yf, ybf);

  // 5. fused FFN + final LN (WMMA)
  ffn_fused_kernel<<<NN_NODES / 16, T, 0, stream>>>(ybf, yf, W1t, W2t, b1, alpha, b2,
                                                    ln_g, ln_b, out);
}
